// GCN_5016521802364
// MI455X (gfx1250) — matrix operand order repaired and checked by reading, not compiled, not run on hardware
//
#include <hip/hip_runtime.h>
#include <stdint.h>

#define NNODES 50000
#define KDEG 16
#define EDGES (NNODES * KDEG)
#define MTILES (NNODES / 16)   // 3125
#define LSTM_BLOCKS 1563       // persistent blocks: each handles at most 2 tiles

typedef __attribute__((ext_vector_type(16))) __bf16 bf16x16;
typedef __attribute__((ext_vector_type(8)))  float  f32x8;
typedef __attribute__((ext_vector_type(4)))  unsigned int v4u;
typedef __attribute__((ext_vector_type(8)))  int  v8i;
typedef __attribute__((ext_vector_type(4)))  int  v4i;

#if defined(__has_builtin)
#if __has_builtin(__builtin_amdgcn_tensor_load_to_lds) && \
    __has_builtin(__builtin_amdgcn_s_wait_tensorcnt)
#define HAVE_TDM 1
#endif
#endif
#ifndef HAVE_TDM
#define HAVE_TDM 0
#endif

struct B32 { uint4 lo, hi; };
static_assert(sizeof(B32) == 32, "B32 must be 32 bytes");
static_assert(sizeof(bf16x16) == 32, "bf16x16 must be 32 bytes");

__device__ __forceinline__ __bf16 f2bf(float f) {
    union { float f; uint32_t u; } v; v.f = f;
    uint32_t r = v.u + 0x7FFFu + ((v.u >> 16) & 1u);   // round-to-nearest-even
    union { uint16_t u; __bf16 b; } o; o.u = (uint16_t)(r >> 16);
    return o.b;
}

// ---- WMMA fragment loaders (row-major bf16 source, ld in elements) -------
// A fragment (16x32): lane holds row M = lane%16; half = lane/16.
// elems 0..7  -> K = k0 + 16*half + 0..7     (16 contiguous bytes)
// elems 8..15 -> K = k0 + 16*half + 8..15    (16 contiguous bytes)
__device__ __forceinline__ bf16x16 load_a_frag(const __bf16* base, int ld,
                                               int row0, int k0, int lane) {
    const int r    = row0 + (lane & 15);
    const int half = lane >> 4;
    const __bf16* p = base + (size_t)r * ld + k0 + half * 16;
    B32 b;
    b.lo = *(const uint4*)(p);
    b.hi = *(const uint4*)(p + 8);    // +16 bytes
    return __builtin_bit_cast(bf16x16, b);
}

// B fragment (32x16): lane holds col N = lane%16; half = lane/16;
// 16 contiguous K values starting at k0 + 16*half.
// W is stored row-major [Nout,K] (i.e. we compute X @ W^T).
__device__ __forceinline__ bf16x16 load_b_frag(const __bf16* base, int ld,
                                               int n0, int k0, int lane) {
    const int n    = n0 + (lane & 15);
    const int half = lane >> 4;
    const __bf16* p = base + (size_t)n * ld + k0 + half * 16;
    B32 b;
    b.lo = *(const uint4*)(p);
    b.hi = *(const uint4*)(p + 8);    // +16 bytes
    return __builtin_bit_cast(bf16x16, b);
}

__device__ __forceinline__ f32x8 wmma_bf16(bf16x16 a, bf16x16 b, f32x8 c) {
    return __builtin_amdgcn_wmma_f32_16x16x32_bf16(
        /*neg_a=*/false, a, /*neg_b=*/false, b,
        /*c_mod=*/(short)0, c, /*reuse_a=*/false, /*reuse_b=*/false);
}

#if HAVE_TDM
// TDM: DMA a row-major 2D bf16 matrix [rows x cols] from global into LDS.
// D# group0: count=1 | lds_addr | global_addr | type=2.
// D# group1: data_size=2B, tensor dims/strides == tile dims (full matrix).
// 6-arg builtin: (uint32x4 g0, int32x8 g1, int32x4, int32x4, int32x8, i32 cpol)
__device__ __forceinline__ void tdm_load_2d_bf16(uint32_t lds_addr,
                                                 const void* gptr,
                                                 uint32_t rows, uint32_t cols) {
    const uint64_t ga = (uint64_t)(uintptr_t)gptr;
    v4u g0;
    g0.x = 1u;                                    // count=1 valid descriptor
    g0.y = lds_addr;                              // LDS byte address
    g0.z = (uint32_t)ga;                          // global_addr[31:0]
    g0.w = (uint32_t)((ga >> 32) & 0x1FFFFFFu) | (2u << 30);  // addr[56:32]|type=2
    v8i g1;
    g1.s0 = 0x00010000;                           // wg_mask=0, data_size=1 (2B)
    g1.s1 = (int)(cols << 16);                    // tensor_dim0[15:0] @bits63:48
    g1.s2 = (int)(rows << 16);                    // tensor_dim1[15:0] @bits95:80
    g1.s3 = (int)(cols << 16);                    // tile_dim0 @bits127:112
    g1.s4 = (int)rows;                            // tile_dim1 @bits143:128
    g1.s5 = (int)cols;                            // tensor_dim0_stride[31:0]
    g1.s6 = 0;
    g1.s7 = 0;
    v4i z4 = {0, 0, 0, 0};
    v8i z8 = {0, 0, 0, 0, 0, 0, 0, 0};
    __builtin_amdgcn_tensor_load_to_lds(g0, g1, z4, z4, z8, 0);
}
#endif

// ---------------------------------------------------------------- convert
__global__ __launch_bounds__(256) void f32_to_bf16_kernel(
        const float* __restrict__ in, __bf16* __restrict__ out, int n) {
    int i = blockIdx.x * 256 + threadIdx.x;
    if (i < n) out[i] = f2bf(in[i]);
}

// -------------------------------------------------- GEMM (+dual) + bias + relu
// Y[M,Nout] = relu(A[M,K] @ W[Nout,K]^T + bias (+ A2[M,K2] @ W2[Nout,K2]^T))
// One wave per 16x16 output tile; 8 waves (tiles along M) per block.
__global__ __launch_bounds__(256) void gemm_bias_relu_kernel(
        const __bf16* __restrict__ A,  const __bf16* __restrict__ W,
        const float*  __restrict__ bias,
        const __bf16* __restrict__ A2, const __bf16* __restrict__ W2,
        __bf16* __restrict__ outB, float* __restrict__ outF,
        int Nout, int K, int K2)
{
    const int lane  = threadIdx.x & 31;
    const int wave  = threadIdx.x >> 5;
    const int mtile = blockIdx.x * 8 + wave;
    if (mtile >= MTILES) return;                     // wave-uniform branch
    const int m0 = mtile << 4;
    const int n0 = blockIdx.y << 4;

    const float bval = bias ? bias[n0 + (lane & 15)] : 0.f;
    f32x8 acc;
#pragma unroll
    for (int i = 0; i < 8; ++i) acc[i] = bval;

    for (int k0 = 0; k0 < K; k0 += 32) {
        bf16x16 a = load_a_frag(A, K, m0, k0, lane);
        bf16x16 b = load_b_frag(W, K, n0, k0, lane);
        acc = wmma_bf16(a, b, acc);
    }
    if (A2) {
        for (int k0 = 0; k0 < K2; k0 += 32) {
            bf16x16 a = load_a_frag(A2, K2, m0, k0, lane);
            bf16x16 b = load_b_frag(W2, K2, n0, k0, lane);
            acc = wmma_bf16(a, b, acc);
        }
    }
#pragma unroll
    for (int r = 0; r < 8; ++r) {
        float v = acc[r];
        v = v > 0.f ? v : 0.f;
        const int m = m0 + r + ((lane >> 4) << 3);   // C layout: M=r(+8), N=lane%16
        const int n = n0 + (lane & 15);
        if (outB) outB[(size_t)m * Nout + n] = f2bf(v);
        if (outF) outF[(size_t)m * Nout + n] = v;
    }
}

// --------------------------------------------------------------- LSTM aggr
// Persistent blocks; each handles 16-node tiles in a grid-stride loop.
// hidden=input=64, 16 timesteps, gates=256.
// Wave w owns gate columns [32w, 32w+32): 2 accumulators, 8 WMMAs/step.
// Weight B-fragments are extracted once into registers (8 x v16bf per wave);
// the recurrent loop touches LDS only for xt/h A-fragments and the gate buffer.
__global__ __launch_bounds__(256) void lstm_aggr_kernel(
        const __bf16* __restrict__ xp,    // [N,64] projected+relu feats (bf16)
        const int*    __restrict__ src,   // [E] neighbor indices
        const __bf16* __restrict__ wih,   // [256,64] bf16
        const __bf16* __restrict__ whh,   // [256,64] bf16
        const float*  __restrict__ bih, const float* __restrict__ bhh,
        __bf16* __restrict__ aggr)        // [N,64] bf16 out
{
    __shared__ __bf16 s_w [256 * 64];     // staging buffer (wih, then whh)
    __shared__ __bf16 s_xt[16 * 64];
    __shared__ __bf16 s_h [16 * 64];
    __shared__ float  s_g [16 * 256];
    __shared__ int    s_idx[256];

    const int tid  = threadIdx.x;
    const int lane = tid & 31;
    const int wave = tid >> 5;
    const int n0a = wave * 32, n0b = n0a + 16;

    // ---- Stage weights via TDM and pin B-fragments in registers ----------
    bf16x16 wx[4], wh[4];                 // [tile a/b][k-chunk]
#if HAVE_TDM
    if (wave == 0) {
        tdm_load_2d_bf16((uint32_t)(uintptr_t)s_w, wih, 256u, 64u);
        __builtin_amdgcn_s_wait_tensorcnt(0);
    }
#else
    for (int i = tid; i < 2048; i += 256)
        ((uint4*)s_w)[i] = ((const uint4*)wih)[i];
#endif
    __syncthreads();
    wx[0] = load_b_frag(s_w, 64, n0a, 0,  lane);
    wx[1] = load_b_frag(s_w, 64, n0a, 32, lane);
    wx[2] = load_b_frag(s_w, 64, n0b, 0,  lane);
    wx[3] = load_b_frag(s_w, 64, n0b, 32, lane);
    __syncthreads();
#if HAVE_TDM
    if (wave == 0) {
        tdm_load_2d_bf16((uint32_t)(uintptr_t)s_w, whh, 256u, 64u);
        __builtin_amdgcn_s_wait_tensorcnt(0);
    }
#else
    for (int i = tid; i < 2048; i += 256)
        ((uint4*)s_w)[i] = ((const uint4*)whh)[i];
#endif
    __syncthreads();
    wh[0] = load_b_frag(s_w, 64, n0a, 0,  lane);
    wh[1] = load_b_frag(s_w, 64, n0a, 32, lane);
    wh[2] = load_b_frag(s_w, 64, n0b, 0,  lane);
    wh[3] = load_b_frag(s_w, 64, n0b, 32, lane);

    const int c0 = n0a + (lane & 15);                 // this wave's gate cols
    const int c1 = c0 + 16;
    const float b0 = bih[c0] + bhh[c0];
    const float b1 = bih[c1] + bhh[c1];
    const bool  isg = ((wave >> 1) == 2);             // cols 128..191 -> tanh gate

    for (int mt = blockIdx.x; mt < MTILES; mt += gridDim.x) {
        const int m0 = mt << 4;

        // Per-tile reset: neighbor indices, h0 = 0, c0 = 0.
        s_idx[tid] = src[m0 * KDEG + tid];            // e = m0*16 + tid
        ((uint2*)s_h)[tid] = make_uint2(0u, 0u);
        float cst[4] = {0.f, 0.f, 0.f, 0.f};
        __syncthreads();

        for (int t = 0; t < KDEG; ++t) {
            // Gather xt rows (16 x 128B) into LDS; prefetch next step's rows.
            {
                const int row = tid >> 4, seg = tid & 15;
                const int s = s_idx[(row << 4) + t];
                ((uint2*)s_xt)[tid] = ((const uint2*)(xp + (size_t)s * 64))[seg];
                if (t + 1 < KDEG) {
                    const int sn = s_idx[(row << 4) + t + 1];
                    __builtin_prefetch(xp + (size_t)sn * 64, 0, 0);
                }
            }
            __syncthreads();

            f32x8 a0, a1;
#pragma unroll
            for (int i = 0; i < 8; ++i) { a0[i] = b0; a1[i] = b1; }

            bf16x16 ax0 = load_a_frag(s_xt, 64, 0, 0,  lane);
            bf16x16 ax1 = load_a_frag(s_xt, 64, 0, 32, lane);
            bf16x16 ah0 = load_a_frag(s_h,  64, 0, 0,  lane);
            bf16x16 ah1 = load_a_frag(s_h,  64, 0, 32, lane);

            a0 = wmma_bf16(ax0, wx[0], a0);
            a0 = wmma_bf16(ax1, wx[1], a0);
            a0 = wmma_bf16(ah0, wh[0], a0);
            a0 = wmma_bf16(ah1, wh[1], a0);
            a1 = wmma_bf16(ax0, wx[2], a1);
            a1 = wmma_bf16(ax1, wx[3], a1);
            a1 = wmma_bf16(ah0, wh[2], a1);
            a1 = wmma_bf16(ah1, wh[3], a1);

            // Apply gate activations and park in LDS.
#pragma unroll
            for (int r = 0; r < 8; ++r) {
                const int nl = r + ((lane >> 4) << 3);    // local node 0..15
                float v0 = a0[r], v1 = a1[r];
                if (isg) { v0 = tanhf(v0); v1 = tanhf(v1); }
                else     { v0 = 1.f / (1.f + expf(-v0));
                           v1 = 1.f / (1.f + expf(-v1)); }
                s_g[nl * 256 + n0a + (lane & 15)] = v0;
                s_g[nl * 256 + n0b + (lane & 15)] = v1;
            }
            __syncthreads();

            // Pointwise c/h update; h back to LDS (bf16) for recurrent matmul.
#pragma unroll
            for (int q = 0; q < 4; ++q) {
                const int p    = tid + (q << 8);
                const int node = p >> 6;
                const int j    = p & 63;
                const float gi = s_g[node * 256 +       j];
                const float gf = s_g[node * 256 +  64 + j];
                const float gg = s_g[node * 256 + 128 + j];
                const float go = s_g[node * 256 + 192 + j];
                const float c  = gf * cst[q] + gi * gg;
                cst[q] = c;
                s_h[node * 64 + j] = f2bf(go * tanhf(c));
            }
            __syncthreads();
        }

        // Write final hidden state (16 x 128B), then fence before reuse.
        ((uint2*)(aggr + (size_t)m0 * 64))[tid] = ((uint2*)s_h)[tid];
        __syncthreads();
    }
}

// =========================================================================
extern "C" void kernel_launch(void* const* d_in, const int* in_sizes, int n_in,
                              void* d_out, int out_size, void* d_ws, size_t ws_size,
                              hipStream_t stream)
{
    const float* x      = (const float*)d_in[0];
    const int*   eidx   = (const int*)d_in[1];          // [2,E]; src = eidx[0:E]
    const float* p1_pw  = (const float*)d_in[3];
    const float* p1_pb  = (const float*)d_in[4];
    const float* p1_wih = (const float*)d_in[5];
    const float* p1_whh = (const float*)d_in[6];
    const float* p1_bih = (const float*)d_in[7];
    const float* p1_bhh = (const float*)d_in[8];
    const float* p1_llw = (const float*)d_in[9];
    const float* p1_llb = (const float*)d_in[10];
    const float* p1_lrw = (const float*)d_in[11];
    const float* p2_pw  = (const float*)d_in[12];
    const float* p2_pb  = (const float*)d_in[13];
    const float* p2_wih = (const float*)d_in[14];
    const float* p2_whh = (const float*)d_in[15];
    const float* p2_bih = (const float*)d_in[16];
    const float* p2_bhh = (const float*)d_in[17];
    const float* p2_llw = (const float*)d_in[18];
    const float* p2_llb = (const float*)d_in[19];
    const float* p2_lrw = (const float*)d_in[20];
    const float* l1w    = (const float*)d_in[21];
    const float* l1b    = (const float*)d_in[22];
    const float* l2w    = (const float*)d_in[23];
    const float* l2b    = (const float*)d_in[24];
    const float* l3w    = (const float*)d_in[25];
    const float* l3b    = (const float*)d_in[26];

    // ---- workspace layout (bf16 buffers) -------------------------------
    size_t off = 0;
    auto take = [&](size_t elems) {
        void* p = (char*)d_ws + off;
        off += ((elems * 2 + 255) & ~(size_t)255);
        return (__bf16*)p;
    };
    __bf16* w_p1p  = take(64 * 64);
    __bf16* w_p1ih = take(256 * 64);
    __bf16* w_p1hh = take(256 * 64);
    __bf16* w_p1ll = take(64 * 64);
    __bf16* w_p1lr = take(64 * 64);
    __bf16* w_p2p  = take(64 * 64);
    __bf16* w_p2ih = take(256 * 64);
    __bf16* w_p2hh = take(256 * 64);
    __bf16* w_p2ll = take(128 * 64);
    __bf16* w_p2lr = take(128 * 64);
    __bf16* w_l1   = take(192 * 128);
    __bf16* w_l2   = take(64 * 192);
    __bf16* w_l3   = take(64 * 64);
    __bf16* B0 = take((size_t)NNODES * 64);
    __bf16* B1 = take((size_t)NNODES * 64);
    __bf16* B2 = take((size_t)NNODES * 64);
    __bf16* B3 = take((size_t)NNODES * 128);
    __bf16* B4 = take((size_t)NNODES * 192);

    auto conv = [&](const float* s, __bf16* d, int n) {
        f32_to_bf16_kernel<<<(n + 255) / 256, 256, 0, stream>>>(s, d, n);
    };
    conv(p1_pw,  w_p1p,  64 * 64);
    conv(p1_wih, w_p1ih, 256 * 64);
    conv(p1_whh, w_p1hh, 256 * 64);
    conv(p1_llw, w_p1ll, 64 * 64);
    conv(p1_lrw, w_p1lr, 64 * 64);
    conv(p2_pw,  w_p2p,  64 * 64);
    conv(p2_wih, w_p2ih, 256 * 64);
    conv(p2_whh, w_p2hh, 256 * 64);
    conv(p2_llw, w_p2ll, 128 * 64);
    conv(p2_lrw, w_p2lr, 128 * 64);
    conv(l1w,    w_l1,   192 * 128);
    conv(l2w,    w_l2,   64 * 192);
    conv(l3w,    w_l3,   64 * 64);
    conv(x, B0, NNODES * 64);                        // x -> bf16

    auto gemm = [&](const __bf16* A, const __bf16* W, const float* bias,
                    const __bf16* A2, const __bf16* W2,
                    __bf16* oB, float* oF, int Nout, int K, int K2) {
        dim3 g((MTILES + 7) / 8, Nout / 16);
        gemm_bias_relu_kernel<<<g, 256, 0, stream>>>(A, W, bias, A2, W2,
                                                     oB, oF, Nout, K, K2);
    };
    auto lstm = [&](const __bf16* xp, const __bf16* wih, const __bf16* whh,
                    const float* bi, const float* bh, __bf16* out) {
        lstm_aggr_kernel<<<LSTM_BLOCKS, 256, 0, stream>>>(xp, eidx, wih, whh,
                                                          bi, bh, out);
    };

    // ------- SAGE layer 1 -----------------------------------------------
    gemm(B0, w_p1p, p1_pb, nullptr, nullptr, B1, nullptr, 64, 64, 0);   // xp = relu(proj(x))
    lstm(B1, w_p1ih, w_p1hh, p1_bih, p1_bhh, B2);                       // aggr
    gemm(B2, w_p1ll, p1_llb, B0, w_p1lr, B1, nullptr, 64, 64, 64);      // h1 = relu(ll+lr)
    // ------- SAGE layer 2 -----------------------------------------------
    gemm(B1, w_p2p, p2_pb, nullptr, nullptr, B0, nullptr, 64, 64, 0);   // xp2
    lstm(B0, w_p2ih, w_p2hh, p2_bih, p2_bhh, B2);                       // aggr2
    gemm(B2, w_p2ll, p2_llb, B1, w_p2lr, B3, nullptr, 128, 64, 64);     // h2 [N,128]
    // ------- MLP ---------------------------------------------------------
    gemm(B3, w_l1, l1b, nullptr, nullptr, B4, nullptr, 192, 128, 0);
    gemm(B4, w_l2, l2b, nullptr, nullptr, B0, nullptr, 64, 192, 0);
    gemm(B0, w_l3, l3b, nullptr, nullptr, nullptr, (float*)d_out, 64, 64, 0);

    // Output tuple: (h [N,64] f32, edge_index [2,E] i32) concatenated.
    (void)hipMemcpyAsync((float*)d_out + (size_t)NNODES * 64, eidx,
                         sizeof(int) * 2 * EDGES, hipMemcpyDeviceToDevice, stream);

    (void)in_sizes; (void)n_in; (void)out_size; (void)ws_size;
}
